// GFM_29180007809275
// MI455X (gfx1250) — matrix operand order repaired and checked by reading, not compiled, not run on hardware
//
#include <hip/hip_runtime.h>
#include <hip/hip_bf16.h>
#include <math.h>

// ---------------------------------------------------------------------------
// CDNA5 (gfx1250) wave32 WMMA implementation of the attention+MLP block.
// All matmuls use v_wmma_f32_16x16x32_bf16 (bf16 in, f32 accumulate).
// Dense GEMMs: 64x64 workgroup tile, 4 waves, 2x2 fragment reuse per wave.
// ---------------------------------------------------------------------------

typedef __attribute__((ext_vector_type(16))) __bf16 v16bf;
typedef __attribute__((ext_vector_type(8)))  __bf16 v8bf;
typedef __attribute__((ext_vector_type(8)))  float  v8f;

#define BB 4
#define CC 384
#define HH 28
#define WW 28
#define NN 784            // H*W
#define NT 3136           // B*N
#define NHEADS 8
#define HDIM 48
#define C2 768            // 2*C
#define SCALE 0.14433756729740643f   // 48^-0.5

__device__ __forceinline__ float gelu_exact(float x) {
    return 0.5f * x * (1.0f + erff(x * 0.70710678118654752f));
}

// Assemble a lane's 16-element WMMA fragment from two contiguous v8bf chunks.
__device__ __forceinline__ v16bf frag_cat(v8bf lo, v8bf hi) {
    return __builtin_shufflevector(lo, hi, 0, 1, 2, 3, 4, 5, 6, 7,
                                          8, 9, 10, 11, 12, 13, 14, 15);
}

// A fragment for lane (row = lane&15, half = lane>>4) from a 32-wide bf16 row.
__device__ __forceinline__ v16bf frag_a(const __bf16* row, int half) {
    const v8bf* p = (const v8bf*)row;
    return frag_cat(p[half], p[2 + half]);
}
// B fragment (col = lane&15, half selects K 8h..8h+7 and 16+8h..16+8h+7).
__device__ __forceinline__ v16bf frag_b(const __bf16* row, int half) {
    const v8bf* p = (const v8bf*)row;
    return frag_cat(p[half], p[2 + half]);
}

// ---------------------------------------------------------------------------
// 1) depthwise 3x3 pos conv + residual + bias; x[B,C,H,W] -> t[B,N,C]
// ---------------------------------------------------------------------------
__global__ __launch_bounds__(256)
void posconv_kernel(const float* __restrict__ x, const float* __restrict__ pw,
                    const float* __restrict__ pb, float* __restrict__ t) {
    int idx = blockIdx.x * 256 + threadIdx.x;
    if (idx >= BB * CC * HH * WW) return;
    int w = idx % WW;
    int h = (idx / WW) % HH;
    int c = (idx / (WW * HH)) % CC;
    int b = idx / (WW * HH * CC);
    const float* xp = x + ((size_t)(b * CC + c)) * (HH * WW);
    float acc = 0.f;
#pragma unroll
    for (int dy = 0; dy < 3; ++dy)
#pragma unroll
        for (int dx = 0; dx < 3; ++dx) {
            int yy = h + dy - 1, xx = w + dx - 1;
            if (yy >= 0 && yy < HH && xx >= 0 && xx < WW)
                acc += xp[yy * WW + xx] * pw[c * 9 + dy * 3 + dx];
        }
    float val = xp[h * WW + w] + acc + pb[c];
    t[((size_t)b * NN + h * WW + w) * CC + c] = val;
}

// ---------------------------------------------------------------------------
// 2) LayerNorm over last dim (C=384), one wave per row, shfl reductions
// ---------------------------------------------------------------------------
__global__ __launch_bounds__(32)
void ln_kernel(const float* __restrict__ in, const float* __restrict__ w,
               const float* __restrict__ bias, float* __restrict__ out) {
    int row = blockIdx.x;
    const float* x = in + (size_t)row * CC;
    float s = 0.f;
    for (int c = threadIdx.x; c < CC; c += 32) s += x[c];
#pragma unroll
    for (int off = 16; off; off >>= 1) s += __shfl_xor(s, off);
    float mean = s * (1.0f / CC);
    float v = 0.f;
    for (int c = threadIdx.x; c < CC; c += 32) {
        float d = x[c] - mean;
        v += d * d;
    }
#pragma unroll
    for (int off = 16; off; off >>= 1) v += __shfl_xor(v, off);
    float rstd = rsqrtf(v * (1.0f / CC) + 1e-6f);
    float* o = out + (size_t)row * CC;
    for (int c = threadIdx.x; c < CC; c += 32)
        o[c] = (x[c] - mean) * rstd * w[c] + bias[c];
}

// ---------------------------------------------------------------------------
// 3) Dense GEMM: out[m][o] = act( A[m][:] . W[o][:] + bias[o] ) + res[m][o]
//    A: [M,K] row-major, W: [Nout,K] row-major.
//    Block = 128 threads (4 waves) -> 64x64 tile; wave -> 32x32, 2x2 frags.
//    Requires M % 64 == 0, Nout % 64 == 0, K % 32 == 0.
// ---------------------------------------------------------------------------
__global__ __launch_bounds__(128)
void gemm_wmma_kernel(const float* __restrict__ A, int K,
                      const float* __restrict__ W,
                      const float* __restrict__ bias,
                      const float* __restrict__ res,
                      float* __restrict__ out, int Nout, int act) {
    __shared__ __bf16 As[64][32];   // [m][k]
    __shared__ __bf16 Bs[64][32];   // [n][k]
    int ntn = Nout >> 6;
    int m0 = (blockIdx.x / ntn) * 64;
    int n0 = (blockIdx.x % ntn) * 64;
    int tid = threadIdx.x;
    int lane = tid & 31;
    int wave = tid >> 5;            // 0..3
    int wm = (wave & 1) * 32;       // wave M offset within 64x64 tile
    int wn = (wave >> 1) * 32;      // wave N offset
    int half = lane >> 4;
    int l15 = lane & 15;
    int lr = tid >> 1;              // 0..63: row loaded by this thread
    int lc = (tid & 1) * 16;        // col group (16 floats, contiguous)

    v8f zero = {0.f, 0.f, 0.f, 0.f, 0.f, 0.f, 0.f, 0.f};
    v8f acc[2][2] = {{zero, zero}, {zero, zero}};

    for (int k0 = 0; k0 < K; k0 += 32) {
        const float* asrc = A + (size_t)(m0 + lr) * K + k0 + lc;
        const float* bsrc = W + (size_t)(n0 + lr) * K + k0 + lc;
        if (k0 + 32 < K) {          // prefetch next K-step (global_prefetch_b8)
            __builtin_prefetch(asrc + 32);
            __builtin_prefetch(bsrc + 32);
        }
#pragma unroll
        for (int i = 0; i < 16; ++i) {
            As[lr][lc + i] = (__bf16)asrc[i];
            Bs[lr][lc + i] = (__bf16)bsrc[i];
        }
        __syncthreads();
        v16bf fa0 = frag_a(&As[wm + l15][0], half);
        v16bf fa1 = frag_a(&As[wm + 16 + l15][0], half);
        v16bf fb0 = frag_b(&Bs[wn + l15][0], half);
        v16bf fb1 = frag_b(&Bs[wn + 16 + l15][0], half);
        acc[0][0] = __builtin_amdgcn_wmma_f32_16x16x32_bf16(
            false, fa0, false, fb0, (short)0, acc[0][0], false, false);
        acc[0][1] = __builtin_amdgcn_wmma_f32_16x16x32_bf16(
            false, fa0, false, fb1, (short)0, acc[0][1], false, false);
        acc[1][0] = __builtin_amdgcn_wmma_f32_16x16x32_bf16(
            false, fa1, false, fb0, (short)0, acc[1][0], false, false);
        acc[1][1] = __builtin_amdgcn_wmma_f32_16x16x32_bf16(
            false, fa1, false, fb1, (short)0, acc[1][1], false, false);
        __syncthreads();
    }
#pragma unroll
    for (int mi = 0; mi < 2; ++mi)
#pragma unroll
        for (int ni = 0; ni < 2; ++ni) {
            int col = n0 + wn + ni * 16 + l15;
            float bv = bias ? bias[col] : 0.f;
#pragma unroll
            for (int rr = 0; rr < 8; ++rr) {
                int m = m0 + wm + mi * 16 + rr + half * 8;
                float v = acc[mi][ni][rr] + bv;
                if (act == 1) v = gelu_exact(v);
                if (res) v += res[(size_t)m * Nout + col];
                out[(size_t)m * Nout + col] = v;
            }
        }
}

// ---------------------------------------------------------------------------
// 4) S[b,h,n,m] = scale * q[b,n,h,:] . k[b,m,h,:]   (K = 48, zero-padded)
// ---------------------------------------------------------------------------
__global__ __launch_bounds__(32)
void attn_qk_kernel(const float* __restrict__ q, const float* __restrict__ kv,
                    float* __restrict__ S) {
    __shared__ __bf16 As[16][32];
    __shared__ __bf16 Bs[16][32];
    int b = blockIdx.z, h = blockIdx.y;
    int mt = blockIdx.x % 49, nt = blockIdx.x / 49;
    int m0 = mt * 16, n0 = nt * 16;
    int lane = threadIdx.x;
    int half = lane >> 4;
    int l15 = lane & 15;
    int r = lane >> 1;
    int c0 = (lane & 1) * 16;

    v8f acc = {0.f, 0.f, 0.f, 0.f, 0.f, 0.f, 0.f, 0.f};
    for (int k0 = 0; k0 < 64; k0 += 32) {
        const float* aq = q + (size_t)(b * NN + n0 + r) * CC + h * HDIM;
        const float* bk = kv + (size_t)(b * NN + m0 + r) * C2 + h * HDIM;
#pragma unroll
        for (int i = 0; i < 16; ++i) {
            int kk = k0 + c0 + i;
            As[r][c0 + i] = (kk < HDIM) ? (__bf16)aq[kk] : (__bf16)0.f;
            Bs[r][c0 + i] = (kk < HDIM) ? (__bf16)bk[kk] : (__bf16)0.f;
        }
        __syncthreads();
        v16bf a = frag_a(&As[l15][0], half);
        v16bf bfr = frag_b(&Bs[l15][0], half);
        acc = __builtin_amdgcn_wmma_f32_16x16x32_bf16(
            false, a, false, bfr, (short)0, acc, false, false);
        __syncthreads();
    }
#pragma unroll
    for (int rr = 0; rr < 8; ++rr) {
        int nrow = n0 + rr + half * 8;
        S[(((size_t)(b * NHEADS + h)) * NN + nrow) * NN + m0 + l15] =
            acc[rr] * SCALE;
    }
}

// ---------------------------------------------------------------------------
// 5) head-mix MLP, pointwise over (b,n,m): 8 -> 32 (gelu) -> 8, in place
// ---------------------------------------------------------------------------
__global__ __launch_bounds__(256)
void headmix_kernel(float* __restrict__ S, const float* __restrict__ fc1_w,
                    const float* __restrict__ fc1_b,
                    const float* __restrict__ fc2_w,
                    const float* __restrict__ fc2_b, long total) {
    __shared__ float w1[256], b1[32], w2[256], b2[8];
    int t = threadIdx.x;
    w1[t] = fc1_w[t];
    w2[t] = fc2_w[t];
    if (t < 32) b1[t] = fc1_b[t];
    if (t < 8) b2[t] = fc2_b[t];
    __syncthreads();
    long idx = (long)blockIdx.x * 256 + t;
    if (idx >= total) return;
    const long NN2 = (long)NN * NN;
    int b = (int)(idx / NN2);
    long rem = idx - (long)b * NN2;
    int n = (int)(rem / NN);
    int m = (int)(rem % NN);
    float s[NHEADS];
#pragma unroll
    for (int h = 0; h < NHEADS; ++h)
        s[h] = S[(((size_t)(b * NHEADS + h)) * NN + n) * NN + m];
    float u[32];
#pragma unroll
    for (int j = 0; j < 32; ++j) {
        float a = b1[j];
#pragma unroll
        for (int h = 0; h < NHEADS; ++h) a += w1[j * 8 + h] * s[h];
        u[j] = gelu_exact(a);
    }
#pragma unroll
    for (int h = 0; h < NHEADS; ++h) {
        float a = b2[h];
#pragma unroll
        for (int j = 0; j < 32; ++j) a += w2[h * 32 + j] * u[j];
        S[(((size_t)(b * NHEADS + h)) * NN + n) * NN + m] = a;
    }
}

// ---------------------------------------------------------------------------
// 6) softmax over last dim (784), one wave per row, in place
// ---------------------------------------------------------------------------
__global__ __launch_bounds__(32)
void softmax_kernel(float* __restrict__ S) {
    float* p = S + (size_t)blockIdx.x * NN;
    float mx = -1e30f;
    for (int i = threadIdx.x; i < NN; i += 32) mx = fmaxf(mx, p[i]);
#pragma unroll
    for (int off = 16; off; off >>= 1) mx = fmaxf(mx, __shfl_xor(mx, off));
    float sum = 0.f;
    for (int i = threadIdx.x; i < NN; i += 32) {
        float e = __expf(p[i] - mx);
        p[i] = e;
        sum += e;
    }
#pragma unroll
    for (int off = 16; off; off >>= 1) sum += __shfl_xor(sum, off);
    float inv = 1.0f / sum;
    for (int i = threadIdx.x; i < NN; i += 32) p[i] *= inv;
}

// ---------------------------------------------------------------------------
// 7) o[b,n,h,:] = P[b,h,n,:] . v[b,:,h,:]   (K = 784, last step zero-padded)
//    One wave -> 16 rows x full 48-dim head: 3 accumulators reuse A fragment.
// ---------------------------------------------------------------------------
__global__ __launch_bounds__(32)
void attn_pv_kernel(const float* __restrict__ P, const float* __restrict__ kv,
                    float* __restrict__ o) {
    __shared__ __bf16 As[16][32];   // [n][k]
    __shared__ __bf16 Bs[48][32];   // [d][k]
    int b = blockIdx.z, h = blockIdx.y;
    int n0 = blockIdx.x * 16;
    int lane = threadIdx.x;
    int half = lane >> 4;
    int l15 = lane & 15;
    int r = lane >> 1;
    int c0 = (lane & 1) * 16;

    v8f zero = {0.f, 0.f, 0.f, 0.f, 0.f, 0.f, 0.f, 0.f};
    v8f acc[3] = {zero, zero, zero};
    const float* ap = P + (((size_t)(b * NHEADS + h)) * NN + n0 + r) * NN;
    const float* vbase = kv + CC + h * HDIM;  // v[b, m, h*48 + d]

    for (int k0 = 0; k0 < NN; k0 += 32) {     // 25 steps; last is zero-padded
#pragma unroll
        for (int i = 0; i < 16; ++i) {
            int kk = k0 + c0 + i;
            As[r][c0 + i] = (kk < NN) ? (__bf16)ap[kk] : (__bf16)0.f;
        }
#pragma unroll
        for (int j = 0; j < 3; ++j) {
            int d = j * 16 + r;
#pragma unroll
            for (int i = 0; i < 16; ++i) {
                int kk = k0 + c0 + i;
                Bs[d][c0 + i] =
                    (kk < NN) ? (__bf16)vbase[(size_t)(b * NN + kk) * C2 + d]
                              : (__bf16)0.f;
            }
        }
        __syncthreads();
        v16bf fa = frag_a(&As[l15][0], half);
#pragma unroll
        for (int j = 0; j < 3; ++j) {
            v16bf fb = frag_b(&Bs[j * 16 + l15][0], half);
            acc[j] = __builtin_amdgcn_wmma_f32_16x16x32_bf16(
                false, fa, false, fb, (short)0, acc[j], false, false);
        }
        __syncthreads();
    }
#pragma unroll
    for (int j = 0; j < 3; ++j)
#pragma unroll
        for (int rr = 0; rr < 8; ++rr) {
            int nrow = n0 + rr + half * 8;
            o[(size_t)(b * NN + nrow) * CC + h * HDIM + j * 16 + l15] =
                acc[j][rr];
        }
}

// ---------------------------------------------------------------------------
// 8) t[B,N,C] -> out[B,C,H,W]
// ---------------------------------------------------------------------------
__global__ __launch_bounds__(256)
void to_nchw_kernel(const float* __restrict__ t, float* __restrict__ out) {
    int idx = blockIdx.x * 256 + threadIdx.x;
    if (idx >= BB * CC * HH * WW) return;
    int hw = idx % (HH * WW);
    int c = (idx / (HH * WW)) % CC;
    int b = idx / (HH * WW * CC);
    out[idx] = t[((size_t)b * NN + hw) * CC + c];
}

// ---------------------------------------------------------------------------
extern "C" void kernel_launch(void* const* d_in, const int* in_sizes, int n_in,
                              void* d_out, int out_size, void* d_ws,
                              size_t ws_size, hipStream_t stream) {
    const float* x      = (const float*)d_in[0];
    const float* pos_w  = (const float*)d_in[1];
    const float* pos_b  = (const float*)d_in[2];
    const float* ln1_w  = (const float*)d_in[3];
    const float* ln1_b  = (const float*)d_in[4];
    const float* q_w    = (const float*)d_in[5];
    const float* q_b    = (const float*)d_in[6];
    const float* kv_w   = (const float*)d_in[7];
    const float* kv_b   = (const float*)d_in[8];
    const float* fc1_w  = (const float*)d_in[9];
    const float* fc1_b  = (const float*)d_in[10];
    const float* fc2_w  = (const float*)d_in[11];
    const float* fc2_b  = (const float*)d_in[12];
    const float* proj_w = (const float*)d_in[13];
    const float* proj_b = (const float*)d_in[14];
    const float* ln2_w  = (const float*)d_in[15];
    const float* ln2_b  = (const float*)d_in[16];
    const float* m1_w   = (const float*)d_in[17];
    const float* m1_b   = (const float*)d_in[18];
    const float* m2_w   = (const float*)d_in[19];
    const float* m2_b   = (const float*)d_in[20];

    float* ws = (float*)d_ws;
    float* t  = ws;                 // NT*C      = 1,204,224 f
    float* xn = t + 1204224;        // NT*C
    float* q  = xn + 1204224;       // NT*C
    float* kv = q + 1204224;        // NT*2C     = 2,408,448 f
    float* S  = kv + 2408448;       // B*H*N*N   = 19,668,992 f
    float* o  = S + 19668992;       // NT*C
    float* y  = o + 1204224;        // NT*4C     = 4,816,896 f

    dim3 b32(32), b128(128), b256(256);
    int elems = BB * CC * HH * WW;

    // pos conv + residual -> tokens
    posconv_kernel<<<dim3((elems + 255) / 256), b256, 0, stream>>>(x, pos_w,
                                                                   pos_b, t);
    // LN1
    ln_kernel<<<dim3(NT), b32, 0, stream>>>(t, ln1_w, ln1_b, xn);
    // q = xn @ q_w^T + q_b            (49 x 6 tiles of 64x64)
    gemm_wmma_kernel<<<dim3(49 * 6), b128, 0, stream>>>(xn, CC, q_w, q_b,
                                                        nullptr, q, CC, 0);
    // kv = xn @ kv_w^T + kv_b         (49 x 12)
    gemm_wmma_kernel<<<dim3(49 * 12), b128, 0, stream>>>(xn, CC, kv_w, kv_b,
                                                         nullptr, kv, C2, 0);
    // S = scale * q k^T
    attn_qk_kernel<<<dim3(49 * 49, NHEADS, BB), b32, 0, stream>>>(q, kv, S);
    // head-mixing MLP (in place)
    long total = (long)BB * NN * NN;
    headmix_kernel<<<dim3((unsigned)((total + 255) / 256)), b256, 0, stream>>>(
        S, fc1_w, fc1_b, fc2_w, fc2_b, total);
    // softmax (in place)
    softmax_kernel<<<dim3(BB * NHEADS * NN), b32, 0, stream>>>(S);
    // o = P @ v                        (one wave per 16 rows x 48 dims)
    attn_pv_kernel<<<dim3(49, NHEADS, BB), b32, 0, stream>>>(S, kv, o);
    // t = t + o @ proj_w^T + proj_b    (in-place residual; 49 x 6)
    gemm_wmma_kernel<<<dim3(49 * 6), b128, 0, stream>>>(o, CC, proj_w, proj_b,
                                                        t, t, CC, 0);
    // LN2
    ln_kernel<<<dim3(NT), b32, 0, stream>>>(t, ln2_w, ln2_b, xn);
    // y = gelu(xn @ m1_w^T + m1_b)     (49 x 24)
    gemm_wmma_kernel<<<dim3(49 * 24), b128, 0, stream>>>(xn, CC, m1_w, m1_b,
                                                         nullptr, y, 4 * CC, 1);
    // t = t + y @ m2_w^T + m2_b        (49 x 6)
    gemm_wmma_kernel<<<dim3(49 * 6), b128, 0, stream>>>(y, 4 * CC, m2_w, m2_b,
                                                        t, t, CC, 0);
    // tokens -> NCHW output
    to_nchw_kernel<<<dim3((elems + 255) / 256), b256, 0, stream>>>(
        t, (float*)d_out);
}
